// TRBlock_30090540876461
// MI455X (gfx1250) — matrix operand order repaired and checked by reading, not compiled, not run on hardware
//
#include <hip/hip_runtime.h>

// ---------------- problem constants ----------------
#define BB 4
#define NN 8192
#define CC 64
#define KK 16
#define TILE 8                    // query points per workgroup
#define COLS (TILE * KK)          // 128 point-neighbor columns per workgroup
#define BN_SCALE 0.9999950000374997f   // 1/sqrt(1+1e-5)

typedef __attribute__((ext_vector_type(16))) _Float16 v16h;
typedef __attribute__((ext_vector_type(8)))  _Float16 v8h;
typedef __attribute__((ext_vector_type(8)))  float    v8f;

// ---------------- WMMA operand loaders ----------------
// A operand: weights W[o][c] (f16, row stride 64) for tile (mt, kc).
// ISA 16-bit A 16x32 layout: lane m=lane&15, kh=lane>>4;
//   elements 0..7  -> K = kh*8 + (0..7)
//   elements 8..15 -> K = 16 + kh*8 + (0..7)
__device__ inline v16h load_wmma_a(const _Float16* w, int mt, int kc, int lane) {
    int m  = lane & 15;
    int kh = lane >> 4;
    int o  = mt * 16 + m;
    int c0 = kc * 32 + kh * 8;
    union { v16h v; v8h h[2]; } u;
    u.h[0] = *(const v8h*)(w + o * 64 + c0);
    u.h[1] = *(const v8h*)(w + o * 64 + c0 + 16);
    return u.v;
}

// B operand: activations act[col][c] (f16, row stride 64).
// ISA 16-bit B 32x16 layout: lane col=lane&15, kh=lane>>4;
//   elements 0..7  -> K = kh*8 + (0..7)
//   elements 8..15 -> K = 16 + kh*8 + (0..7)
__device__ inline v16h load_wmma_b(const _Float16* act, int tile_base, int kc, int lane) {
    int n   = lane & 15;
    int kb  = (lane >> 4) * 8;
    int col = tile_base + n;
    int c0  = kc * 32 + kb;
    union { v16h v; v8h h[2]; } u;
    u.h[0] = *(const v8h*)(act + col * 64 + c0);
    u.h[1] = *(const v8h*)(act + col * 64 + c0 + 16);
    return u.v;
}

// Stage one 8KB f16 weight matrix global->LDS on the async path (ASYNCcnt).
// Each of the 256 threads moves 2x16B. Caller must s_wait_asynccnt + barrier
// before the weights are read, and only issue after a barrier that retired
// all readers of sW.
__device__ inline void stage_weights_async(const _Float16* gsrc, _Float16* sW, int t) {
    unsigned lds0 = (unsigned)(uintptr_t)sW + (unsigned)t * 16u;
    unsigned off0 = (unsigned)t * 16u;
    unsigned long long base = (unsigned long long)(uintptr_t)gsrc;
    asm volatile("global_load_async_to_lds_b128 %0, %1, %2 offset:0"
                 :: "v"(lds0), "v"(off0), "s"(base) : "memory");
    asm volatile("global_load_async_to_lds_b128 %0, %1, %2 offset:4096"
                 :: "v"(lds0 + 4096u), "v"(off0), "s"(base) : "memory");
}

__device__ inline void wait_async_lds() {
    asm volatile("s_wait_asynccnt 0x0" ::: "memory");
}

// One fused GEMM layer: dst[col][o] = act( mul[o]*(W*src)[o][col] + add[o] (+ gx) )
// All 4 M-tiles kept in independent accumulators so WMMAs pipeline without
// dependent-chain hazard NOPs; B operands loaded once.
// C/D layout: lane col=lane&15, hi=lane>>4; element r -> channel mt*16 + hi*8 + r.
// NOTE: gx may alias dst (same-thread read-then-write of identical slots).
__device__ inline void gemm_layer(const _Float16* __restrict__ src,
                                  _Float16* __restrict__ dst,
                                  const _Float16* __restrict__ w,
                                  const float* __restrict__ mulv,
                                  const float* __restrict__ addv,
                                  const _Float16* __restrict__ gx,
                                  int lane, int tile_base, bool relu) {
    v16h b0 = load_wmma_b(src, tile_base, 0, lane);
    v16h b1 = load_wmma_b(src, tile_base, 1, lane);
    v8f acc[4];
#pragma unroll
    for (int mt = 0; mt < 4; ++mt) {
        v16h a0 = load_wmma_a(w, mt, 0, lane);
        v16h a1 = load_wmma_a(w, mt, 1, lane);
        v8f c = {};
        c = __builtin_amdgcn_wmma_f32_16x16x32_f16(false, a0, false, b0,
                                                   (short)0, c, false, false);
        acc[mt] = __builtin_amdgcn_wmma_f32_16x16x32_f16(false, a1, false, b1,
                                                         (short)0, c, false, false);
    }
    const int hi  = lane >> 4;
    const int n   = lane & 15;
    const int col = tile_base + n;
#pragma unroll
    for (int mt = 0; mt < 4; ++mt) {
        const int ch0 = mt * 16 + hi * 8;
        v8h gxv = {};
        if (gx) gxv = *(const v8h*)(gx + col * 64 + ch0);
        v8h outv;
#pragma unroll
        for (int r = 0; r < 8; ++r) {
            const int ch = ch0 + r;
            float v = acc[mt][r] * mulv[ch] + addv[ch];
            if (gx)   v += (float)gxv[r];
            if (relu) v  = v > 0.f ? v : 0.f;
            outv[r] = (_Float16)v;
        }
        *(v8h*)(dst + col * 64 + ch0) = outv;
    }
}

// ---------------- fused point-transformer block ----------------
__global__ void __launch_bounds__(256)
ptblock_kernel(const float* __restrict__ p, const float* __restrict__ xin,
               const int* __restrict__ idx, const _Float16* __restrict__ wh16,
               const float* dw1, const float* db1, const float* dg1, const float* dbe1,
               const float* db2, const float* dg2, const float* dbe2,
               const float* ab1, const float* ag1, const float* abe1,
               const float* ab2, const float* ag2, const float* abe2,
               const float* lb,  const float* lg,  const float* lbe,
               float* __restrict__ xout, int blk) {
    __shared__ __align__(16) _Float16 sW[64 * 64];      // current-layer weights, f16 [o][c]
    __shared__ __align__(16) _Float16 sA0[COLS * 64];   // activations ping  [col][c]
    __shared__ __align__(16) _Float16 sA1[COLS * 64];   // activations pong  [col][c] (also gx)
    __shared__ float sPar[5][2][64];                    // [layer][mul/add][ch]
    __shared__ float sDw1[64 * 3];

    const int t    = threadIdx.x;
    const int lane = t & 31;
    const int wave = t >> 5;
    const int b    = blockIdx.x / (NN / TILE);
    const int n0   = (blockIdx.x % (NN / TILE)) * TILE;

    const _Float16* Wd2h = wh16 + (size_t)(blk * 4 + 0) * 4096;
    const _Float16* Wa1h = wh16 + (size_t)(blk * 4 + 1) * 4096;
    const _Float16* Wa2h = wh16 + (size_t)(blk * 4 + 2) * 4096;
    const _Float16* Wdnh = wh16 + (size_t)(blk * 4 + 3) * 4096;

    // ---- phase 0: async-stage dw2; fold BN into per-channel mul/add ----
    stage_weights_async(Wd2h, sW, t);
    if (t < 64) {
        const int o = blk * CC + t;
        float m;
        m = dg1[o] * BN_SCALE; sPar[0][0][t] = m; sPar[0][1][t] = db1[o] * m + dbe1[o];
        m = dg2[o] * BN_SCALE; sPar[1][0][t] = m; sPar[1][1][t] = db2[o] * m + dbe2[o];
        m = ag1[o] * BN_SCALE; sPar[2][0][t] = m; sPar[2][1][t] = ab1[o] * m + abe1[o];
        m = ag2[o] * BN_SCALE; sPar[3][0][t] = m; sPar[3][1][t] = ab2[o] * m + abe2[o];
        m = lg[o]  * BN_SCALE; sPar[4][0][t] = m; sPar[4][1][t] = lb[o]  * m + lbe[o];
    }
    if (t < 192) sDw1[t] = dw1[blk * CC * 3 + t];
    __syncthreads();

    // ---- phase 1: gather rel/gx + delta layer-1 (64x3, VALU) ----
    // gx goes straight into sA1 (the pe destination buffer) -- the pe epilogue
    // reads each slot before overwriting it from the same thread.
    {
        const int col  = t & (COLS - 1);     // two threads per column
        const int half = t >> 7;             // channel half 0/1
        const int pi   = col >> 4;
        const int k    = col & 15;
        const int n    = n0 + pi;
        const int j    = idx[(b * NN + n) * KK + k];
        const float* pb = p + (size_t)b * 3 * NN;
        const float r0 = pb[n]          - pb[j];
        const float r1 = pb[NN + n]     - pb[NN + j];
        const float r2 = pb[2 * NN + n] - pb[2 * NN + j];
        const float* xb = xin + (size_t)b * CC * NN;
        const int cbeg = half * 32;
#pragma unroll 4
        for (int c = cbeg; c < cbeg + 32; ++c)
            sA1[col * 64 + c] = (_Float16)xb[(size_t)c * NN + j];
#pragma unroll 4
        for (int o = cbeg; o < cbeg + 32; ++o) {
            float v = sDw1[o * 3] * r0 + sDw1[o * 3 + 1] * r1 + sDw1[o * 3 + 2] * r2;
            v = v * sPar[0][0][o] + sPar[0][1][o];
            v = v > 0.f ? v : 0.f;
            sA0[col * 64 + o] = (_Float16)v;
        }
    }
    wait_async_lds();
    __syncthreads();

    // ---- phase 2: pe = bn(dw2 * h); t = gx + pe  (sA0 -> sA1, gx aliased) ----
    gemm_layer(sA0, sA1, sW, sPar[1][0], sPar[1][1], sA1, lane, wave * 16, false);
    __syncthreads();

    // ---- phase 3: alpha layer 1 (relu)  (sA1 -> sA0) ----
    stage_weights_async(Wa1h, sW, t);
    wait_async_lds();
    __syncthreads();
    gemm_layer(sA1, sA0, sW, sPar[2][0], sPar[2][1], nullptr, lane, wave * 16, true);
    __syncthreads();

    // ---- phase 4: alpha layer 2 (relu)  (sA0 -> sA1) ----
    stage_weights_async(Wa2h, sW, t);
    wait_async_lds();
    __syncthreads();
    gemm_layer(sA0, sA1, sW, sPar[3][0], sPar[3][1], nullptr, lane, wave * 16, true);
    __syncthreads();

    // ---- phase 5: stage down_w (async, overlaps with K-max); max over K ----
    stage_weights_async(Wdnh, sW, t);
    for (int e = t; e < TILE * 64; e += 256) {
        const int pt = e >> 6;
        const int c  = e & 63;
        float mx = (float)sA1[(pt * KK) * 64 + c];
#pragma unroll
        for (int k = 1; k < KK; ++k) {
            float v = (float)sA1[(pt * KK + k) * 64 + c];
            mx = v > mx ? v : mx;
        }
        sA0[pt * 64 + c] = (_Float16)mx;
    }
    wait_async_lds();
    __syncthreads();

    // ---- phase 6: down GEMM (64x16) + bn + residual ----
    if (wave < 4) {
        const int mt = wave;
        v8f acc = {};
        v16h a0 = load_wmma_a(sW, mt, 0, lane);
        v16h b0 = load_wmma_b(sA0, 0, 0, lane);
        acc = __builtin_amdgcn_wmma_f32_16x16x32_f16(false, a0, false, b0,
                                                     (short)0, acc, false, false);
        v16h a1 = load_wmma_a(sW, mt, 1, lane);
        v16h b1 = load_wmma_b(sA0, 0, 1, lane);
        acc = __builtin_amdgcn_wmma_f32_16x16x32_f16(false, a1, false, b1,
                                                     (short)0, acc, false, false);
        const int hi = lane >> 4;
        const int n  = lane & 15;
        if (n < TILE) {
            const int ch0 = mt * 16 + hi * 8;
            const int gn  = n0 + n;
#pragma unroll
            for (int r = 0; r < 8; ++r) {
                const int ch = ch0 + r;
                float v = acc[r] * sPar[4][0][ch] + sPar[4][1][ch];
                const size_t ga = (size_t)b * CC * NN + (size_t)ch * NN + gn;
                xout[ga] = v + xin[ga];
            }
        }
    }
}

// ---------------- one-time weight conversion f32 -> f16 ----------------
// Layout: [blk(2)][layer: dw2, aw1, aw2, down_w][64*64]
__global__ void cvt_weights_kernel(const float* __restrict__ dw2, const float* __restrict__ aw1,
                                   const float* __restrict__ aw2, const float* __restrict__ lw,
                                   _Float16* __restrict__ out) {
    const int i = blockIdx.x * blockDim.x + threadIdx.x;   // 0 .. 32767
    const int blk   = i >> 14;
    const int r     = i & 16383;
    const int layer = r >> 12;
    const int e     = r & 4095;
    const float* src = (layer == 0) ? dw2 : (layer == 1) ? aw1 : (layer == 2) ? aw2 : lw;
    out[i] = (_Float16)src[blk * 4096 + e];
}

// ---------------- passthrough copy of positions ----------------
__global__ void copy_p_kernel(const float* __restrict__ src, float* __restrict__ dst, int n) {
    int i = blockIdx.x * blockDim.x + threadIdx.x;
    if (i < n) dst[i] = src[i];
}

extern "C" void kernel_launch(void* const* d_in, const int* in_sizes, int n_in,
                              void* d_out, int out_size, void* d_ws, size_t ws_size,
                              hipStream_t stream) {
    const float* p   = (const float*)d_in[0];
    const float* x   = (const float*)d_in[1];
    const int*   idx = (const int*)d_in[2];
    const float* dw1 = (const float*)d_in[3];
    const float* db1 = (const float*)d_in[4];
    const float* dg1 = (const float*)d_in[5];
    const float* dbe1= (const float*)d_in[6];
    const float* dw2 = (const float*)d_in[7];
    const float* db2 = (const float*)d_in[8];
    const float* dg2 = (const float*)d_in[9];
    const float* dbe2= (const float*)d_in[10];
    const float* aw1 = (const float*)d_in[11];
    const float* ab1 = (const float*)d_in[12];
    const float* ag1 = (const float*)d_in[13];
    const float* abe1= (const float*)d_in[14];
    const float* aw2 = (const float*)d_in[15];
    const float* ab2 = (const float*)d_in[16];
    const float* ag2 = (const float*)d_in[17];
    const float* abe2= (const float*)d_in[18];
    const float* lw  = (const float*)d_in[19];
    const float* lb  = (const float*)d_in[20];
    const float* lg  = (const float*)d_in[21];
    const float* lbe = (const float*)d_in[22];

    float* out   = (float*)d_out;
    float* x_mid = (float*)d_ws;                               // B*C*N floats intermediate
    _Float16* wh16 = (_Float16*)((char*)d_ws + (size_t)BB * CC * NN * 4);  // 8 matrices f16
    float* x_out = out + (size_t)BB * 3 * NN;                  // output x region after p

    // p passes through unchanged
    const int pElems = BB * 3 * NN;
    copy_p_kernel<<<(pElems + 255) / 256, 256, 0, stream>>>(p, out, pElems);

    // one-time f32 -> f16 weight conversion (2 blocks x 4 layers x 64x64)
    cvt_weights_kernel<<<(2 * 4 * 4096) / 256, 256, 0, stream>>>(dw2, aw1, aw2, lw, wh16);

    const dim3 grid(BB * (NN / TILE));
    // block 0: input_x -> workspace
    ptblock_kernel<<<grid, 256, 0, stream>>>(p, x, idx, wh16,
        dw1, db1, dg1, dbe1, db2, dg2, dbe2,
        ab1, ag1, abe1, ab2, ag2, abe2,
        lb, lg, lbe, x_mid, 0);
    // block 1: workspace -> output
    ptblock_kernel<<<grid, 256, 0, stream>>>(p, x_mid, idx, wh16,
        dw1, db1, dg1, dbe1, db2, dg2, dbe2,
        ab1, ag1, abe1, ab2, ag2, abe2,
        lb, lg, lbe, x_out, 1);
}
